// Step3p5MoEMLP_7687991460209
// MI455X (gfx1250) — matrix operand order repaired and checked by reading, not compiled, not run on hardware
//
#include <hip/hip_runtime.h>
#include <hip/hip_bf16.h>

#define T_TOK 1024
#define H_DIM 2048
#define I_DIM 1024
#define E_NUM 32

typedef __attribute__((ext_vector_type(16))) __bf16 v16bf;
typedef __attribute__((ext_vector_type(8)))  float  v8f;

__device__ __forceinline__ __bf16 f2bf(float f) { return (__bf16)f; }

__device__ __forceinline__ unsigned short f2bfbits(float f) {
    union { __bf16 b; unsigned short s; } c;
    c.b = (__bf16)f;
    return c.s;
}

// Convert 8 contiguous floats (two float4) into bf16 elements [base..base+7] of a v16bf.
__device__ __forceinline__ void cvt8(v16bf& r, int base, const float4 a, const float4 b) {
    r[base + 0] = f2bf(a.x); r[base + 1] = f2bf(a.y);
    r[base + 2] = f2bf(a.z); r[base + 3] = f2bf(a.w);
    r[base + 4] = f2bf(b.x); r[base + 5] = f2bf(b.y);
    r[base + 6] = f2bf(b.z); r[base + 7] = f2bf(b.w);
}

// Load 16 contiguous floats and convert into a v16bf (B-fragment helper).
__device__ __forceinline__ v16bf load_cvt16(const float* p) {
    v16bf r;
    cvt8(r, 0, ((const float4*)p)[0], ((const float4*)p)[1]);
    cvt8(r, 8, ((const float4*)p)[2], ((const float4*)p)[3]);
    return r;
}

// ---------------------------------------------------------------------------
// Kernel 0: zero the output accumulator (we atomicAdd into it) and counters.
// ---------------------------------------------------------------------------
__global__ __launch_bounds__(256) void zero_out(float* __restrict__ out,
                                                int* __restrict__ counts) {
    size_t i = (size_t)blockIdx.x * 256 + threadIdx.x;
    float4 z; z.x = 0.f; z.y = 0.f; z.z = 0.f; z.w = 0.f;
    ((float4*)out)[i] = z;
    if (i < E_NUM) counts[i] = 0;
}

// ---------------------------------------------------------------------------
// Kernel 1: router. One wave (32 lanes) per token; lane = expert.
// ---------------------------------------------------------------------------
__global__ __launch_bounds__(256) void router(const float* __restrict__ x,
                                              const float* __restrict__ gw,
                                              int*   __restrict__ counts,
                                              int*   __restrict__ tok,
                                              float* __restrict__ wgt) {
    const int lane = threadIdx.x & 31;
    const int wave = threadIdx.x >> 5;
    const int t = blockIdx.x * 8 + wave;           // 8 tokens per block

    const float* xr = x  + (size_t)t    * H_DIM;
    const float* wr = gw + (size_t)lane * H_DIM;   // expert = lane

    float acc = 0.f;
    for (int h = 0; h < H_DIM; h += 4) {
        float4 xv = *(const float4*)(xr + h);
        float4 wv = *(const float4*)(wr + h);
        acc += xv.x * wv.x + xv.y * wv.y + xv.z * wv.z + xv.w * wv.w;
    }

    // wave32 softmax
    float m = acc;
    for (int off = 16; off; off >>= 1) m = fmaxf(m, __shfl_xor(m, off, 32));
    float p = __expf(acc - m);
    float s = p;
    for (int off = 16; off; off >>= 1) s += __shfl_xor(s, off, 32);
    float prob = p / s;

    // top-1 (value, index)
    float v0 = prob; int i0 = lane;
    for (int off = 16; off; off >>= 1) {
        float ov = __shfl_xor(v0, off, 32);
        int   oi = __shfl_xor(i0, off, 32);
        if (ov > v0 || (ov == v0 && oi < i0)) { v0 = ov; i0 = oi; }
    }
    // top-2 (exclude i0)
    float v1 = (lane == i0) ? -1.f : prob; int i1 = lane;
    for (int off = 16; off; off >>= 1) {
        float ov = __shfl_xor(v1, off, 32);
        int   oi = __shfl_xor(i1, off, 32);
        if (ov > v1 || (ov == v1 && oi < i1)) { v1 = ov; i1 = oi; }
    }

    float inv = 1.f / (v0 + v1);    // ROUTED_SCALING == 1.0
    if (lane == 0) {
        int s0 = atomicAdd(&counts[i0], 1);
        tok[i0 * T_TOK + s0] = t * 2 + 0;
        wgt[i0 * T_TOK + s0] = v0 * inv;
        int s1 = atomicAdd(&counts[i1], 1);
        tok[i1 * T_TOK + s1] = t * 2 + 1;
        wgt[i1 * T_TOK + s1] = v1 * inv;
    }
}

// ---------------------------------------------------------------------------
// Kernel 2: gate/up projections + SwiGLU + combine weight, per expert.
// grid = (T/16 token tiles, I/256 col tiles, E). block = 256 (8 waves);
// each wave owns a 16x32 output tile (two 16x16 N-tiles sharing one A
// fragment): 4x v_wmma_f32_16x16x32_bf16 per K-step over H.
// ---------------------------------------------------------------------------
__global__ __launch_bounds__(256) void moe_gu(const float* __restrict__ x,
                                              const float* __restrict__ gpw,
                                              const float* __restrict__ upw,
                                              const int*   __restrict__ counts,
                                              const int*   __restrict__ tok,
                                              const float* __restrict__ wgt,
                                              unsigned short* __restrict__ hbuf) {
    const int e  = blockIdx.z;
    const int ne = counts[e];
    const int m0 = blockIdx.x << 4;
    if (m0 >= ne) return;

    const int lane  = threadIdx.x & 31;
    const int wave  = threadIdx.x >> 5;
    const int half  = lane >> 4;
    const int mlane = lane & 15;
    const int nG0   = (blockIdx.y << 8) + (wave << 5) + mlane;  // first col tile
    const int nG1   = nG0 + 16;                                 // second col tile

    int slot = m0 + mlane; if (slot > ne - 1) slot = ne - 1;
    const int eT  = e * T_TOK;
    const int aid = tok[eT + slot];                             // t*2+k
    const float* xr  = x + (size_t)(aid >> 1) * H_DIM;
    const float* wg0 = gpw + ((size_t)e * I_DIM + nG0) * H_DIM;
    const float* wg1 = gpw + ((size_t)e * I_DIM + nG1) * H_DIM;
    const float* wu0 = upw + ((size_t)e * I_DIM + nG0) * H_DIM;
    const float* wu1 = upw + ((size_t)e * I_DIM + nG1) * H_DIM;

    v8f accg0 = {0.f,0.f,0.f,0.f,0.f,0.f,0.f,0.f};
    v8f accg1 = accg0, accu0 = accg0, accu1 = accg0;

    for (int k = 0; k < H_DIM; k += 32) {
        // A fragment: lane row m=mlane; elems 0..15 -> K = k+16*half+0..15
        // (contiguous, matching B).
        v16bf a;
        {
            const float* pa = xr + k + 16 * half;
            cvt8(a, 0, ((const float4*)pa)[0], ((const float4*)pa)[1]);
            cvt8(a, 8, ((const float4*)(pa + 8))[0], ((const float4*)(pa + 8))[1]);
        }
        // B fragments: lane col n; K = k + 16*half + 0..15 contiguous in the
        // weight row (B[k][n] = W[n][k]).
        const int ko = k + 16 * half;
        v16bf bg0 = load_cvt16(wg0 + ko);
        v16bf bg1 = load_cvt16(wg1 + ko);
        v16bf bu0 = load_cvt16(wu0 + ko);
        v16bf bu1 = load_cvt16(wu1 + ko);

        accg0 = __builtin_amdgcn_wmma_f32_16x16x32_bf16(false, a, false, bg0,
                                                        (short)0, accg0, false, false);
        accg1 = __builtin_amdgcn_wmma_f32_16x16x32_bf16(false, a, false, bg1,
                                                        (short)0, accg1, false, false);
        accu0 = __builtin_amdgcn_wmma_f32_16x16x32_bf16(false, a, false, bu0,
                                                        (short)0, accu0, false, false);
        accu1 = __builtin_amdgcn_wmma_f32_16x16x32_bf16(false, a, false, bu1,
                                                        (short)0, accu1, false, false);
    }

    // D layout: VGPR r holds row m = r + 8*half, col n = mlane (+ tile base).
    for (int r = 0; r < 8; ++r) {
        int s2 = m0 + r + 8 * half;
        if (s2 < ne) {
            float w    = wgt[eT + s2];
            int   aid2 = tok[eT + s2];
            float g0 = accg0[r], u0 = accu0[r];
            float g1 = accg1[r], u1 = accu1[r];
            // silu(g)*u*combine with fast reciprocal (epilogue only)
            float h0 = g0 * __builtin_amdgcn_rcpf(1.f + __expf(-g0)) * u0 * w;
            float h1 = g1 * __builtin_amdgcn_rcpf(1.f + __expf(-g1)) * u1 * w;
            unsigned short* hp = hbuf + (size_t)aid2 * I_DIM;
            hp[nG0] = f2bfbits(h0);
            hp[nG1] = f2bfbits(h1);
        }
    }
}

// ---------------------------------------------------------------------------
// Kernel 3: down projection, accumulate into out[t] with f32 atomics.
// grid = (T/16 token tiles, H/256 col tiles, E). Each wave: 16x32 tile.
// ---------------------------------------------------------------------------
__global__ __launch_bounds__(256) void moe_down(const unsigned short* __restrict__ hbuf,
                                                const float* __restrict__ dpw,
                                                const int*   __restrict__ counts,
                                                const int*   __restrict__ tok,
                                                float* __restrict__ out) {
    const int e  = blockIdx.z;
    const int ne = counts[e];
    const int m0 = blockIdx.x << 4;
    if (m0 >= ne) return;

    const int lane  = threadIdx.x & 31;
    const int wave  = threadIdx.x >> 5;
    const int half  = lane >> 4;
    const int mlane = lane & 15;
    const int nG0   = (blockIdx.y << 8) + (wave << 5) + mlane;  // H col tile 0
    const int nG1   = nG0 + 16;                                 // H col tile 1

    int slot = m0 + mlane; if (slot > ne - 1) slot = ne - 1;
    const int eT  = e * T_TOK;
    const int aid = tok[eT + slot];
    const unsigned short* ar = hbuf + (size_t)aid * I_DIM;
    const float* wd0 = dpw + ((size_t)e * H_DIM + nG0) * I_DIM;
    const float* wd1 = dpw + ((size_t)e * H_DIM + nG1) * I_DIM;

    v8f acc0 = {0.f,0.f,0.f,0.f,0.f,0.f,0.f,0.f};
    v8f acc1 = acc0;

    union AU { uint4 q[2]; v16bf v; };

    for (int k = 0; k < I_DIM; k += 32) {
        AU a;
        a.q[0] = *(const uint4*)(ar + k + 16 * half);        // bf16 already
        a.q[1] = *(const uint4*)(ar + k + 16 * half + 8);
        const int ko = k + 16 * half;
        v16bf b0 = load_cvt16(wd0 + ko);
        v16bf b1 = load_cvt16(wd1 + ko);
        acc0 = __builtin_amdgcn_wmma_f32_16x16x32_bf16(false, a.v, false, b0,
                                                       (short)0, acc0, false, false);
        acc1 = __builtin_amdgcn_wmma_f32_16x16x32_bf16(false, a.v, false, b1,
                                                       (short)0, acc1, false, false);
    }

    for (int r = 0; r < 8; ++r) {
        int s2 = m0 + r + 8 * half;
        if (s2 < ne) {
            int aid2 = tok[eT + s2];
            float* op = out + (size_t)(aid2 >> 1) * H_DIM;
            atomicAdd(op + nG0, acc0[r]);
            atomicAdd(op + nG1, acc1[r]);
        }
    }
}

// ---------------------------------------------------------------------------
extern "C" void kernel_launch(void* const* d_in, const int* in_sizes, int n_in,
                              void* d_out, int out_size, void* d_ws, size_t ws_size,
                              hipStream_t stream) {
    const float* x   = (const float*)d_in[0];
    const float* gw  = (const float*)d_in[1];
    const float* gpw = (const float*)d_in[2];
    const float* upw = (const float*)d_in[3];
    const float* dpw = (const float*)d_in[4];
    float* out = (float*)d_out;

    // workspace layout
    char* ws = (char*)d_ws;
    int*   counts = (int*)ws;                                    // 32 ints
    int*   tok    = (int*)(ws + 256);                            // E*T ints
    float* wgt    = (float*)(ws + 256 + (size_t)E_NUM * T_TOK * 4);
    unsigned short* hbuf =
        (unsigned short*)(ws + 256 + (size_t)E_NUM * T_TOK * 8); // T*2 x I bf16

    zero_out<<<(T_TOK * H_DIM / 4) / 256, 256, 0, stream>>>(out, counts);
    router<<<T_TOK / 8, 256, 0, stream>>>(x, gw, counts, tok, wgt);
    moe_gu<<<dim3(T_TOK / 16, I_DIM / 256, E_NUM), 256, 0, stream>>>(
        x, gpw, upw, counts, tok, wgt, hbuf);
    moe_down<<<dim3(T_TOK / 16, H_DIM / 256, E_NUM), 256, 0, stream>>>(
        hbuf, dpw, counts, tok, out);
}
